// RecurrentKAN_54159537603106
// MI455X (gfx1250) — matrix operand order repaired and checked by reading, not compiled, not run on hardware
//
#include <hip/hip_runtime.h>
#include <hip/hip_bf16.h>

// ---------------------------------------------------------------------------
// RecurrentKAN on gfx1250 (MI455X): persistent per-(direction, batch-tile)
// workgroups, all matmuls via v_wmma_f32_16x16x32_bf16 (f32 accumulate).
// Round 3: pair-wise (cos,sin)->dword staging with __sincosf, bounded unroll,
// reduced register footprint to stay under 256 VGPRs (avoid s_set_vgpr_msb
// thrash seen in round 2).
// ---------------------------------------------------------------------------

#define BB 128
#define TT 256
#define DD 32
#define HH 64
#define GG 50
#define N3H 192          // 3*H    (12 N-tiles of 16)
#define KT_F 200         // Fourier K-tiles (K = 2*H*G = 6400)
#define CHUNK_KT 10      // K-tiles per LDS angle chunk (double buffered)
#define NCHUNK 20

// packed-bf16 workspace layout (in ushort elements)
#define SZ_Z   (2*12*3*512)      // 36864   [dir][nt][kt][512]
#define OFF_A  SZ_Z              // 36864
#define SZ_A   (2*4*3*512)       // 12288
#define OFF_O  (OFF_A + SZ_A)    // 49152
#define SZ_O   (2*4*2*512)       // 8192
#define OFF_F  (OFF_O + SZ_O)    // 57344
#define SZ_F   (2*4*200*512)     // 819200
#define N_PK   (OFF_F + SZ_F)    // 876544 ushorts = 1,753,088 bytes
#define HF_BYTE_OFF (N_PK * 2)   // float hfinal[2][128][64] follows (64 KB)

typedef __attribute__((ext_vector_type(16))) __bf16 v16bf;
typedef __attribute__((ext_vector_type(8)))  float  v8f;

__device__ __forceinline__ unsigned int f2bf(float f) {
    unsigned int u = __float_as_uint(f);
    unsigned int r = u + 0x7FFFu + ((u >> 16) & 1u);   // round-to-nearest-even
    return r >> 16;
}
__device__ __forceinline__ unsigned int pack2bf(float lo, float hi) {
    return f2bf(lo) | (f2bf(hi) << 16);
}

__device__ __forceinline__ v16bf load_frag(const unsigned short* p) {
    union { uint4 u[2]; v16bf v; } t;
    const uint4* q = (const uint4*)p;
    t.u[0] = q[0];
    t.u[1] = q[1];
    return t.v;
}

__device__ __forceinline__ v8f wmma_bf16(v16bf a, v16bf b, v8f c) {
    return __builtin_amdgcn_wmma_f32_16x16x32_bf16(
        /*neg_a=*/false, a, /*neg_b=*/false, b,
        /*c_mod=*/(short)0, c, /*reuse_a=*/false, /*reuse_b=*/false);
}

__device__ __forceinline__ float sigm(float x)  { return 1.0f / (1.0f + __expf(-x)); }
__device__ __forceinline__ float ftanh(float x) {
    x = fminf(15.0f, fmaxf(-15.0f, x));
    float e = __expf(2.0f * x);
    return (e - 1.0f) / (e + 1.0f);
}

// A-fragment (16x32 bf16) element mapping: lane l, element e -> k-within-tile.
// For e = 2j: k is even, and (e, e+1) -> (k, k+1) share one dword of the frag.
__device__ __forceinline__ int afrag_k(int l, int e) {
    return ((l >> 4) * 16) + e;
}

// ---------------------------------------------------------------------------
// Pack kernel: convert all B-matrices to bf16 wave32 B-fragment layout.
// B-fragment (32x16): lane l, element e -> k = kt*32 + (l>>4)*16 + e,
//                                          n = nt*16 + (l&15)
// ---------------------------------------------------------------------------
__global__ __launch_bounds__(256) void rkan_pack(
    const float* __restrict__ kern,      // (2,D,3H)
    const float* __restrict__ rkern,     // (2,H,3H)
    const float* __restrict__ sub_in,    // (2,D,H)
    const float* __restrict__ sub_state, // (2,H,H)
    const float* __restrict__ four_cos,  // (2,H,H,G)
    const float* __restrict__ four_sin,  // (2,H,H,G)
    const float* __restrict__ agg_w,     // (2,H,H)
    unsigned short* __restrict__ pk)
{
    int idx = blockIdx.x * 256 + threadIdx.x;
    if (idx >= N_PK) return;
    float val;
    if (idx < OFF_A) {                       // b_z [2][12][3][512] : [x|h] -> gates
        int r = idx;
        int d  = r / 18432; r -= d * 18432;
        int nt = r / 1536;  r -= nt * 1536;
        int kt = r / 512;   int q = r - kt * 512;
        int l = q >> 4, e = q & 15;
        int n = nt * 16 + (l & 15);
        int k = kt * 32 + ((l >> 4) * 16) + e;
        val = (k < DD) ? kern[(d * DD + k) * N3H + n]
                       : rkern[(d * HH + (k - DD)) * N3H + n];
    } else if (idx < OFF_O) {                // b_a [2][4][3][512] : [x|s] -> agg
        int r = idx - OFF_A;
        int d  = r / 6144;  r -= d * 6144;
        int nt = r / 1536;  r -= nt * 1536;
        int kt = r / 512;   int q = r - kt * 512;
        int l = q >> 4, e = q & 15;
        int n = nt * 16 + (l & 15);
        int k = kt * 32 + ((l >> 4) * 16) + e;
        val = (k < DD) ? sub_in[(d * DD + k) * HH + n]
                       : sub_state[(d * HH + (k - DD)) * HH + n];
    } else if (idx < OFF_F) {                // b_o [2][4][2][512] : ko -> o-gate
        int r = idx - OFF_O;
        int d  = r / 4096;  r -= d * 4096;
        int nt = r / 1024;  r -= nt * 1024;
        int kt = r / 512;   int q = r - kt * 512;
        int l = q >> 4, e = q & 15;
        int n = nt * 16 + (l & 15);
        int k = kt * 32 + ((l >> 4) * 16) + e;
        val = agg_w[(d * HH + k) * HH + n];
    } else {                                 // b_f [2][4][200][512] : Fourier coeffs
        int r = idx - OFF_F;
        int d  = r / 409600; r -= d * 409600;
        int nt = r / 102400; r -= nt * 102400;
        int kt = r / 512;    int q = r - kt * 512;
        int l = q >> 4, e = q & 15;
        int n  = nt * 16 + (l & 15);         // output feature o
        int kk = kt * 32 + ((l >> 4) * 16) + e;  // 0..6399
        int cb = kk & 1;                     // 0 = cos, 1 = sin
        int p  = kk >> 1;                    // (i,g) pair
        int i  = p / GG;
        int g  = p - i * GG;
        const float* F = cb ? four_sin : four_cos;
        val = F[((d * HH + n) * HH + i) * GG + g];
    }
    pk[idx] = (unsigned short)f2bf(val);
}

// ---------------------------------------------------------------------------
// Main persistent recurrence kernel. grid = 16 (2 dirs x 8 batch tiles of 16),
// block = 256 threads = 8 waves.
// ---------------------------------------------------------------------------
__global__ __launch_bounds__(256, 1) void rkan_main(
    const float* __restrict__ x,         // (B,T,D)
    const float* __restrict__ bias,      // (2,3H)
    const float* __restrict__ four_bias, // (2,H)
    const float* __restrict__ sub_mix,   // (2,2)
    const float* __restrict__ agg_b,     // (2,H)
    const unsigned short* __restrict__ pk,
    float* __restrict__ hfinal)          // (2,128,64)
{
    const int dir = blockIdx.x >> 3;
    const int bt  = blockIdx.x & 7;
    const int tid  = threadIdx.x;
    const int wave = tid >> 5;
    const int lane = tid & 31;

    __shared__ float sh_h[16 * HH];
    __shared__ float sh_c[16 * HH];
    __shared__ float sh_s[16 * HH];
    __shared__ float sh_z[16 * N3H];
    __shared__ float sh_agg[16 * HH];
    __shared__ float sh_ko[16 * HH];
    __shared__ __align__(32) unsigned short A_zh[3 * 512];
    __shared__ __align__(32) unsigned short A_xs[3 * 512];
    __shared__ __align__(32) unsigned short A_f[2][CHUNK_KT * 512];  // double buffer
    __shared__ __align__(32) unsigned short A_ko[2 * 512];

    // zero-init state
    for (int r = tid; r < 16 * HH; r += 256) {
        sh_h[r] = 0.0f; sh_c[r] = 0.0f; sh_s[r] = 0.0f;
    }

    const float sm0 = sub_mix[dir * 2 + 0];
    const float sm1 = sub_mix[dir * 2 + 1];
    const unsigned short* pz = pk + dir * 18432;
    const unsigned short* pa = pk + OFF_A + dir * 6144;
    const unsigned short* po = pk + OFF_O + dir * 4096;
    const unsigned short* pf = pk + OFF_F + dir * 409600;

    // ---- time-invariant per-wave tile setup & register-cached weights ----
    const int n_c  = lane & 15;          // C-frag column within N-tile
    const int mb_c = (lane >> 4) * 8;    // C-frag first row

    v16bf wz[2][3];                      // gate/agg B fragments (this wave's tiles)
    const unsigned short* AfSel[2];
    float* dstB[2];
    int    rsB[2];
    float  biasB[2];
#pragma unroll
    for (int half = 0; half < 2; ++half) {
        int tt2 = wave + half * 8;       // 0..15: tiles 0-11 = z, 12-15 = agg
        bool isagg = (tt2 >= 12);
        int nt = isagg ? (tt2 - 12) : tt2;
        const unsigned short* Bb = isagg ? (pa + nt * 1536) : (pz + nt * 1536);
#pragma unroll
        for (int kt = 0; kt < 3; ++kt)
            wz[half][kt] = load_frag(Bb + kt * 512 + lane * 16);
        AfSel[half] = isagg ? A_xs : A_zh;
        int col = nt * 16 + n_c;
        dstB[half]  = isagg ? (sh_agg + col) : (sh_z + col);
        rsB[half]   = isagg ? HH : N3H;
        biasB[half] = isagg ? 0.0f : bias[dir * N3H + col];
    }

    const float ob_bias = agg_b[dir * HH + (wave & 3) * 16 + n_c];

    float koinit[4];                     // four_bias values for Phase C
#pragma unroll
    for (int r = 0; r < 4; ++r)
        koinit[r] = four_bias[dir * HH + ((tid * 4 + r) & 63)];

    const int nt_f  = wave >> 1;                 // Fourier N-tile 0..3
    const int ktoff = (wave & 1) * (CHUNK_KT / 2);

    unsigned int* A_f0 = (unsigned int*)&A_f[0][0];
    unsigned int* A_f1 = (unsigned int*)&A_f[1][0];

    __syncthreads();

    for (int t = 0; t < TT; ++t) {
        const int te = dir ? (TT - 1 - t) : t;

        // --- Phase A: stage A_zh = [xt|h], A_xs = [xt|s] (paired dwords) ---
        if (t + 1 < TT && tid < 16) {    // prefetch next x-tile row into caches
            int tn = dir ? (TT - 2 - t) : (t + 1);
            __builtin_prefetch(&x[((size_t)(bt * 16 + tid) * TT + tn) * DD], 0, 0);
        }
#pragma unroll
        for (int r = 0; r < 3; ++r) {
            int idx2 = tid * 3 + r;                // dword index 0..767
            int kt = idx2 >> 8;
            int li = (idx2 >> 3) & 31;
            int e  = (idx2 & 7) * 2;
            int m  = li & 15;
            int k  = kt * 32 + afrag_k(li, e);     // even; pair (k, k+1)
            float z0, z1, s0v, s1v;
            if (k < DD) {
                const float* xr = &x[((size_t)(bt * 16 + m) * TT + te) * DD + k];
                z0 = xr[0]; z1 = xr[1]; s0v = z0; s1v = z1;
            } else {
                z0  = sh_h[m * HH + (k - DD)];
                z1  = sh_h[m * HH + (k - DD) + 1];
                s0v = sh_s[m * HH + (k - DD)];
                s1v = sh_s[m * HH + (k - DD) + 1];
            }
            ((unsigned int*)A_zh)[idx2] = pack2bf(z0, z1);
            ((unsigned int*)A_xs)[idx2] = pack2bf(s0v, s1v);
        }
        __syncthreads();

        // --- Phase B: z = [x|h]W_z + b (12 N-tiles), agg = [x|s]W_a (4 N-tiles)
#pragma unroll
        for (int half = 0; half < 2; ++half) {
            v8f acc = {};
#pragma unroll
            for (int kt = 0; kt < 3; ++kt) {
                v16bf a = load_frag(AfSel[half] + kt * 512 + lane * 16);
                acc = wmma_bf16(a, wz[half][kt], acc);
            }
            float* dst = dstB[half];
            int    rs  = rsB[half];
            float  ba  = biasB[half];
#pragma unroll
            for (int v = 0; v < 8; ++v)
                dst[(mb_c + v) * rs] = acc[v] + ba;   // branchless scatter
        }
        __syncthreads();

        // --- Phase C: cell update c = f*c + i*g ; init ko with four_bias ---
#pragma unroll
        for (int r = 0; r < 4; ++r) {
            int idx = tid * 4 + r;                 // 0..1023
            int m = idx >> 6, n = idx & 63;
            float zi = sigm(sh_z[m * N3H + n]);
            float zf = sigm(sh_z[m * N3H + HH + n]);
            float zg = ftanh(sh_z[m * N3H + 2 * HH + n]);
            sh_c[idx] = zf * sh_c[idx] + zi * zg;
            sh_ko[idx] = koinit[r];
        }
        __syncthreads();

        // --- Phase D: Fourier GEMM  ko += [cos|sin](agg*k) @ Fcoef ---
        // K = 6400 interleaved cos/sin: fragment dword j of lane l holds
        // (cos,sin) of ONE angle -> one decode + one __sincosf + one ds_store
        // per dword. Double-buffered chunks: WMMA on buf[ch&1] overlaps
        // staging of chunk ch+1 (XDL co-executes with TRANS/VALU).
        v8f accf = {};
        {
            // prologue: stage chunk 0
#pragma unroll 2
            for (int r = 0; r < 10; ++r) {
                int idx2 = tid * 10 + r;           // dword 0..2559
                int kt = idx2 >> 8;
                int li = (idx2 >> 3) & 31;
                int e  = (idx2 & 7) * 2;
                int m  = li & 15;
                int p  = kt * 16 + (afrag_k(li, e) >> 1);   // (i,g) pair index
                int i  = p / GG;
                int g  = p - i * GG;
                float ang = sh_agg[m * HH + i] * (float)(g + 1);
                float sv, cv;
                __sincosf(ang, &sv, &cv);
                A_f0[idx2] = pack2bf(cv, sv);
            }
            __syncthreads();
            for (int ch = 0; ch < NCHUNK; ++ch) {
                const unsigned short* bufc = A_f[ch & 1];
                const unsigned short* Bc = pf + (size_t)nt_f * KT_F * 512
                                              + (size_t)(ch * CHUNK_KT + ktoff) * 512;
#pragma unroll
                for (int kt = 0; kt < CHUNK_KT / 2; ++kt) {
                    v16bf a = load_frag(bufc + (ktoff + kt) * 512 + lane * 16);
                    v16bf b = load_frag(Bc + kt * 512 + lane * 16);
                    accf = wmma_bf16(a, b, accf);
                }
                if (ch + 1 < NCHUNK) {
                    unsigned int* bufn = ((ch + 1) & 1) ? A_f1 : A_f0;
                    int pbase = (ch + 1) * (CHUNK_KT * 16);
#pragma unroll 2
                    for (int r = 0; r < 10; ++r) {
                        int idx2 = tid * 10 + r;
                        int kt = idx2 >> 8;
                        int li = (idx2 >> 3) & 31;
                        int e  = (idx2 & 7) * 2;
                        int m  = li & 15;
                        int p  = pbase + kt * 16 + (afrag_k(li, e) >> 1);
                        int i  = p / GG;
                        int g  = p - i * GG;
                        float ang = sh_agg[m * HH + i] * (float)(g + 1);
                        float sv, cv;
                        __sincosf(ang, &sv, &cv);
                        bufn[idx2] = pack2bf(cv, sv);
                    }
                }
                __syncthreads();
            }
        }
        {   // reduce the two K-half partials per N-tile via LDS float atomics
            int n = nt_f * 16 + n_c;
#pragma unroll
            for (int v = 0; v < 8; ++v)
                atomicAdd(&sh_ko[(mb_c + v) * HH + n], accf[v]);
        }
        __syncthreads();

        // --- Phase E: sub-state update  s = sm0*ko + sm1*s ; stage A_ko ---
#pragma unroll
        for (int r = 0; r < 4; ++r) {
            int idx = tid * 4 + r;                 // 0..1023
            float ko = sh_ko[idx];
            sh_s[idx] = sm0 * ko + sm1 * sh_s[idx];
        }
#pragma unroll
        for (int r = 0; r < 2; ++r) {
            int idx2 = tid * 2 + r;                // dword 0..511
            int kt = idx2 >> 8;
            int li = (idx2 >> 3) & 31;
            int e  = (idx2 & 7) * 2;
            int m2 = li & 15;
            int k2 = kt * 32 + afrag_k(li, e);     // even; pair (k2, k2+1)
            ((unsigned int*)A_ko)[idx2] =
                pack2bf(sh_ko[m2 * HH + k2], sh_ko[m2 * HH + k2 + 1]);
        }
        __syncthreads();

        // --- Phase F: o = sigmoid(ko @ Aw + Ab); h = o * tanh(c) ---
        if (wave < 4) {
            int nt = wave & 3;
            v8f acc = {};
#pragma unroll
            for (int kt = 0; kt < 2; ++kt) {
                v16bf a = load_frag(A_ko + kt * 512 + lane * 16);
                v16bf b = load_frag(po + nt * 1024 + kt * 512 + lane * 16);
                acc = wmma_bf16(a, b, acc);
            }
            int n = nt * 16 + n_c;
#pragma unroll
            for (int v = 0; v < 8; ++v) {
                int m = mb_c + v;
                float o = sigm(acc[v] + ob_bias);
                sh_h[m * HH + n] = o * ftanh(sh_c[m * HH + n]);
            }
        }
        __syncthreads();
    }

    // write final hidden state
#pragma unroll
    for (int r = 0; r < 4; ++r) {
        int idx = tid * 4 + r;
        int m = idx >> 6, n = idx & 63;
        hfinal[((size_t)dir * BB + bt * 16 + m) * HH + n] = sh_h[m * HH + n];
    }
}

// ---------------------------------------------------------------------------
// Head: out[b] = concat(h_fwd, h_bwd)[b] . head_w + head_b
// ---------------------------------------------------------------------------
__global__ __launch_bounds__(128) void rkan_head(
    const float* __restrict__ hf,        // (2,128,64)
    const float* __restrict__ head_w,    // (128,1)
    const float* __restrict__ head_b,    // (1,)
    float* __restrict__ out)             // (128,)
{
    int b = threadIdx.x;
    float acc = head_b[0];
    for (int j = 0; j < 2 * HH; ++j) {
        float f = hf[((size_t)(j >> 6) * BB + b) * HH + (j & 63)];
        acc += f * head_w[j];
    }
    out[b] = acc;
}

extern "C" void kernel_launch(void* const* d_in, const int* in_sizes, int n_in,
                              void* d_out, int out_size, void* d_ws, size_t ws_size,
                              hipStream_t stream) {
    (void)in_sizes; (void)n_in; (void)out_size; (void)ws_size;
    const float* x          = (const float*)d_in[0];
    const float* kern       = (const float*)d_in[1];
    const float* rkern      = (const float*)d_in[2];
    const float* bias       = (const float*)d_in[3];
    const float* sub_in     = (const float*)d_in[4];
    const float* sub_state  = (const float*)d_in[5];
    const float* four_cos   = (const float*)d_in[6];
    const float* four_sin   = (const float*)d_in[7];
    const float* four_bias  = (const float*)d_in[8];
    const float* sub_mix    = (const float*)d_in[9];
    const float* agg_w      = (const float*)d_in[10];
    const float* agg_b      = (const float*)d_in[11];
    const float* head_w     = (const float*)d_in[12];
    const float* head_b     = (const float*)d_in[13];

    unsigned short* pk = (unsigned short*)d_ws;                 // ~1.71 MB bf16 packed B
    float* hf = (float*)((char*)d_ws + HF_BYTE_OFF);            // 64 KB final hidden

    rkan_pack<<<(N_PK + 255) / 256, 256, 0, stream>>>(
        kern, rkern, sub_in, sub_state, four_cos, four_sin, agg_w, pk);

    rkan_main<<<16, 256, 0, stream>>>(x, bias, four_bias, sub_mix, agg_b, pk, hf);

    rkan_head<<<1, 128, 0, stream>>>(hf, head_w, head_b, (float*)d_out);
}
